// VQVAEQuantize_88313117540727
// MI455X (gfx1250) — matrix operand order repaired and checked by reading, not compiled, not run on hardware
//
#include <hip/hip_runtime.h>

// ---------------------------------------------------------------------------
// VQ-VAE vector quantization for MI455X (gfx1250, wave32, WMMA + async-LDS).
//
// argmin_k dist(n,k) == argmin_k [ 0.5*||e_k||^2 - x_n.e_k ]
//   * ||e_k||^2 exact in fp32 (tiny precompute)
//   * dot via f16 hi/lo split: x.e ~= xh.eh + xh.el + xl.eh
//     (3x V_WMMA_F32_16X16X32_F16 per k-step, ~22-bit effective mantissa)
//
// B (codebook) tiles are staged into LDS ONCE per workgroup with
// GLOBAL_LOAD_ASYNC_TO_LDS_B128 (ASYNCcnt, double-buffered) instead of each
// of the 8 waves redundantly pulling the same 16KB through L0. LDS rows are
// padded to 528B so half-wave ds_load_b128 reads cover all 64 banks.
// ---------------------------------------------------------------------------

typedef __attribute__((ext_vector_type(16))) _Float16 v16h;
typedef __attribute__((ext_vector_type(8)))  _Float16 v8h;
typedef __attribute__((ext_vector_type(8)))  float    v8f;

constexpr int Bn = 16, Cd = 256, Hh = 32, Ww = 32;
constexpr int Nn = Bn * Hh * Ww;            // 16384 spatial positions
constexpr int Kk = 8192;                    // codebook entries
constexpr int Dd = 256;                     // embedding dim
constexpr int ZQ_ELEMS = Bn * Cd * Hh * Ww; // 4194304

constexpr int ROW_B   = Dd * 2;             // 512B global row (f16)
constexpr int LROW_B  = 528;                // padded LDS row (bank-conflict-free)
constexpr int HALF_B  = 16 * LROW_B;        // 8448B per hi/lo region
constexpr int TILE_B  = 2 * HALF_B;         // 16896B per staged tile
constexpr int NTILES  = Kk / 16;            // 512 code tiles

// ---------------------------------------------------------------------------
// Kernel 1: codebook prep. Per code: f16 hi/lo split + exact 0.5*||e||^2.
// Also zeroes the diff accumulator slot in d_out.
// ---------------------------------------------------------------------------
__global__ __launch_bounds__(256) void k_prep_embed(
    const float* __restrict__ w, _Float16* __restrict__ eh, _Float16* __restrict__ el,
    float* __restrict__ nrm_half, float* __restrict__ diff_slot)
{
    int k = blockIdx.x, t = threadIdx.x;
    float x = w[k * Dd + t];
    _Float16 h = (_Float16)x;
    _Float16 l = (_Float16)(x - (float)h);
    eh[k * Dd + t] = h;
    el[k * Dd + t] = l;

    __shared__ float red[256];
    red[t] = x * x;
    __syncthreads();
    for (int s = 128; s > 0; s >>= 1) {
        if (t < s) red[t] += red[t + s];
        __syncthreads();
    }
    if (t == 0) nrm_half[k] = 0.5f * red[0];
    if (k == 0 && t == 0) *diff_slot = 0.0f;
}

// ---------------------------------------------------------------------------
// Kernel 2: transpose z_e [B,C,H,W] -> flat [N,D] with f16 hi/lo split.
// Coalesced reads; scattered writes absorbed by the 192MB L2.
// ---------------------------------------------------------------------------
__global__ __launch_bounds__(256) void k_prep_flat(
    const float* __restrict__ ze, _Float16* __restrict__ fh, _Float16* __restrict__ fl)
{
    int i = blockIdx.x * 256 + threadIdx.x;
    int w = i & 31, h = (i >> 5) & 31, c = (i >> 10) & 255, b = i >> 18;
    int n = (b << 10) + (h << 5) + w;
    float x = ze[i];
    _Float16 hi = (_Float16)x;
    _Float16 lo = (_Float16)(x - (float)hi);
    fh[n * Dd + c] = hi;
    fl[n * Dd + c] = lo;
}

// ---------------------------------------------------------------------------
// Async 16B copy global -> LDS (gfx1250 GLOBAL_LOAD_ASYNC_TO_LDS_B128).
// ---------------------------------------------------------------------------
__device__ __forceinline__ void async_cp16(unsigned lds_off, const char* g)
{
    asm volatile("global_load_async_to_lds_b128 %0, %1, off"
                 :: "v"(lds_off), "v"(g) : "memory");
}
__device__ __forceinline__ void wait_async0()
{
    asm volatile("s_wait_asynccnt 0x0" ::: "memory");
}

// ---------------------------------------------------------------------------
// Kernel 3: GEMM-with-argmin. 256 threads = 8 waves; each wave owns a 16-row
// M-tile with all 8 A k-fragments (hi+lo, f16, 128 VGPRs) resident. The WG
// loops over 512 code tiles; each tile's B data (16 codes x 256d x hi/lo =
// 16KB) is async-staged into LDS once (double-buffered, ASYNCcnt), then each
// wave reads fragments via ds_load_b128 and issues 24 WMMAs + argmin VALU.
// Fragment layouts per ISA 7.12.2:
//   A lane(m=L%16): elements at d = kk*32 + l16*8 .. +7 and +16..+23
//   B lane(n=L%16): elements at d = kk*32 + l16*8 .. +7 and +16..+23
//   C elem v -> (m = v + 8*l16, n = L%16)
// ---------------------------------------------------------------------------
__global__ __launch_bounds__(256) void k_dist_argmin(
    const _Float16* __restrict__ fa_hi, const _Float16* __restrict__ fa_lo,
    const _Float16* __restrict__ eb_hi, const _Float16* __restrict__ eb_lo,
    const float*  __restrict__ nrm_half,
    int* __restrict__ ind_i32, float* __restrict__ ind_out)
{
    __shared__ __align__(16) char stage[2 * TILE_B];   // 33792B double buffer
    __shared__ float s_best[8][16][16];
    __shared__ int   s_idx [8][16][16];

    const int tid   = threadIdx.x;
    const int lane  = tid & 31;
    const int wave  = tid >> 5;
    const int laneN = lane & 15;
    const int l16   = lane >> 4;
    const int mrow  = blockIdx.x * 128 + wave * 16 + laneN;

    // --- A fragments resident in VGPRs (8 k-steps x hi/lo) -----------------
    v16h a_hi[8], a_lo[8];
    {
        const _Float16* ah = fa_hi + (size_t)mrow * Dd;
        const _Float16* al = fa_lo + (size_t)mrow * Dd;
#pragma unroll
        for (int kk = 0; kk < 8; ++kk) {
            int o0 = kk * 32 + l16 * 8;
            v8h h0 = *(const v8h*)(ah + o0);
            v8h h1 = *(const v8h*)(ah + o0 + 16);
            v8h l0 = *(const v8h*)(al + o0);
            v8h l1 = *(const v8h*)(al + o0 + 16);
            a_hi[kk] = __builtin_shufflevector(h0, h1, 0,1,2,3,4,5,6,7,8,9,10,11,12,13,14,15);
            a_lo[kk] = __builtin_shufflevector(l0, l1, 0,1,2,3,4,5,6,7,8,9,10,11,12,13,14,15);
        }
    }

    // Generic->LDS address: low 32 bits of a shared pointer are the LDS offset.
    const unsigned stage_base = (unsigned)(size_t)(&stage[0]);
    const char* gh = (const char*)eb_hi;
    const char* gl = (const char*)eb_lo;

    // Stage one 16KB code tile into LDS buffer `buf` (4 async b128 per thread).
    auto stage_tile = [&](int buf, int t) {
        int k0 = t << 4;
#pragma unroll
        for (int j = 0; j < 4; ++j) {
            int c    = tid + j * 256;        // 0..1023 16B chunks
            int r    = c >> 9;               // 0 = hi region, 1 = lo region
            int code = (c >> 5) & 15;
            int part = c & 31;               // 16B chunk within 512B row
            unsigned loff = stage_base + buf * TILE_B + r * HALF_B
                          + code * LROW_B + part * 16;
            const char* g = (r ? gl : gh) + (size_t)(k0 + code) * ROW_B + part * 16;
            async_cp16(loff, g);
        }
    };

    float best[8];
    int   bidx[8];
#pragma unroll
    for (int v = 0; v < 8; ++v) { best[v] = 3.4e38f; bidx[v] = 0; }

    stage_tile(0, 0);
    wait_async0();
    __syncthreads();

    const unsigned frag_off = laneN * LROW_B + l16 * 16;  // lane's byte offset

    for (int t = 0; t < NTILES; ++t) {
        const int cur = t & 1;
        if (t + 1 < NTILES) stage_tile(cur ^ 1, t + 1);

        const char* bb = stage + cur * TILE_B;
        v8f acc = {};
#pragma unroll
        for (int kk = 0; kk < 8; ++kk) {
            const v8h* ph = (const v8h*)(bb + frag_off + kk * 64);
            const v8h* pl = (const v8h*)(bb + HALF_B + frag_off + kk * 64);
            v8h h0 = ph[0], h1 = ph[2];
            v8h l0 = pl[0], l1 = pl[2];
            v16h bh = __builtin_shufflevector(h0, h1, 0,1,2,3,4,5,6,7,8,9,10,11,12,13,14,15);
            v16h bl = __builtin_shufflevector(l0, l1, 0,1,2,3,4,5,6,7,8,9,10,11,12,13,14,15);
            acc = __builtin_amdgcn_wmma_f32_16x16x32_f16(false, a_hi[kk], false, bh,
                                                         (short)0, acc, false, false);
            acc = __builtin_amdgcn_wmma_f32_16x16x32_f16(false, a_hi[kk], false, bl,
                                                         (short)0, acc, false, false);
            acc = __builtin_amdgcn_wmma_f32_16x16x32_f16(false, a_lo[kk], false, bh,
                                                         (short)0, acc, false, false);
        }

        const int   k0   = t << 4;
        const float nh   = nrm_half[k0 + laneN];
        const int   kIdx = k0 + laneN;
#pragma unroll
        for (int v = 0; v < 8; ++v) {
            float s = nh - acc[v];
            bool  c = s < best[v];
            best[v] = c ? s : best[v];
            bidx[v] = c ? kIdx : bidx[v];
        }

        wait_async0();      // my async chunk of tile t+1 landed
        __syncthreads();    // everyone finished reading buf[cur] + staging done
    }

    // --- Cross-lane reduction: 16 column candidates per row ----------------
#pragma unroll
    for (int v = 0; v < 8; ++v) {
        int m = v + l16 * 8;
        s_best[wave][m][laneN] = best[v];
        s_idx [wave][m][laneN] = bidx[v];
    }
    __syncthreads();
    if (tid < 128) {
        int w2 = tid >> 4, m = tid & 15;
        float bm = s_best[w2][m][0];
        int   bi = s_idx [w2][m][0];
#pragma unroll
        for (int n = 1; n < 16; ++n) {
            float c  = s_best[w2][m][n];
            int   ci = s_idx [w2][m][n];
            if (c < bm || (c == bm && ci < bi)) { bm = c; bi = ci; }
        }
        int row = blockIdx.x * 128 + w2 * 16 + m;
        ind_i32[row] = bi;
        ind_out[row] = (float)bi;   // indices section of d_out
    }
}

// ---------------------------------------------------------------------------
// Kernel 4: gather z_q (== z_q_st values), write transposed [B,C,H,W], and
// accumulate diff = 12.5 * mean((z_q - z_e)^2) via per-block f32 atomic.
// ---------------------------------------------------------------------------
__global__ __launch_bounds__(256) void k_gather_out(
    const float* __restrict__ ze, const float* __restrict__ w,
    const int* __restrict__ ind, float* __restrict__ out,
    float* __restrict__ diff_slot)
{
    int n = blockIdx.x, c = threadIdx.x;
    int b = n >> 10, h = (n >> 5) & 31, ww = n & 31;
    int idx = ind[n];
    float e = w[(size_t)idx * Dd + c];
    int gi = (((b << 8) + c) << 10) + (h << 5) + ww;  // ((b*256+c)*32+h)*32+w
    float z = ze[gi];
    out[gi] = e;
    float d = e - z;

    __shared__ float red[256];
    red[c] = d * d;
    __syncthreads();
    for (int s = 128; s > 0; s >>= 1) {
        if (c < s) red[c] += red[c + s];
        __syncthreads();
    }
    if (c == 0)
        atomicAdd(diff_slot, red[0] * (12.5f / (float)ZQ_ELEMS));
}

// ---------------------------------------------------------------------------
extern "C" void kernel_launch(void* const* d_in, const int* in_sizes, int n_in,
                              void* d_out, int out_size, void* d_ws, size_t ws_size,
                              hipStream_t stream)
{
    const float* z_e     = (const float*)d_in[0];   // [16,256,32,32]
    const float* embed_w = (const float*)d_in[1];   // [8192,256]
    float* out = (float*)d_out;                     // [z_q_st | diff | ind]

    // Workspace carve-out (~24.1 MB total).
    char* ws = (char*)d_ws;
    _Float16* fh  = (_Float16*)(ws);                                          // 8 MB
    _Float16* fl  = (_Float16*)(ws + (size_t)Nn * Dd * 2);                    // 8 MB
    _Float16* eh  = (_Float16*)(ws + (size_t)Nn * Dd * 4);                    // 4 MB
    _Float16* el  = (_Float16*)(ws + (size_t)Nn * Dd * 4 + (size_t)Kk * Dd * 2); // 4 MB
    float*    nrm = (float*)(ws + (size_t)Nn * Dd * 4 + (size_t)Kk * Dd * 4); // 32 KB
    int*      ind = (int*)((char*)nrm + (size_t)Kk * 4);                      // 64 KB

    float* diff_slot = out + ZQ_ELEMS;       // scalar loss
    float* ind_f     = out + ZQ_ELEMS + 1;   // indices (as float values)

    k_prep_embed <<<Kk,           256, 0, stream>>>(embed_w, eh, el, nrm, diff_slot);
    k_prep_flat  <<<ZQ_ELEMS/256, 256, 0, stream>>>(z_e, fh, fl);
    k_dist_argmin<<<Nn / 128,     256, 0, stream>>>(fh, fl, eh, el, nrm, ind, ind_f);
    k_gather_out <<<Nn,           256, 0, stream>>>(z_e, embed_w, ind, out, diff_slot);
}
